// FusedExpertGroup_65180423685437
// MI455X (gfx1250) — matrix operand order repaired and checked by reading, not compiled, not run on hardware
//
#include <hip/hip_runtime.h>
#include <stdint.h>

// Problem constants (fixed by the reference harness)
#define BATCH 8
#define CIN   256
#define HH    64
#define WW    64
#define NEXP  8
#define COUT  256
#define TOPK  2
#define NG    8          // conv groups == GroupNorm groups
#define CING  (CIN/NG)   // 32 input channels per group
#define COUTG (COUT/NG)  // 32 output channels per group
#define HW    (HH*WW)    // 4096
#define EPSV  1e-5f

// Spatial tiling for the implicit-GEMM conv
#define ROWS     4             // image rows per workgroup tile
#define TILE_PIX (ROWS*WW)     // 256 pixels per workgroup
#define XR (ROWS+2)            // haloed rows  (6)
#define XC (WW+2)              // haloed cols  (66)
#define PADW 66                // padded image width/height (64 + 2 halo)
#define WTILE (9*COUTG*CING)   // 9216 bf16 weight elements per (z,g)

typedef __bf16 bf16_t;
typedef __attribute__((ext_vector_type(16))) bf16_t v16bf;
typedef __attribute__((ext_vector_type(8)))  float  v8f;
typedef __attribute__((ext_vector_type(4)))  uint32_t v4u;
typedef __attribute__((ext_vector_type(8)))  int      v8i_t;
typedef __attribute__((ext_vector_type(4)))  int      v4i_t;

union Frag { v16bf v; uint32_t u[8]; };

// ---------------------------------------------------------------------------
// Tensor Data Mover: build a D# (groups 0..3 per CDNA5 ISA
// section 8) and issue tensor_load_to_lds (6-arg clang-23 form: the extra
// int32x8 operand corresponds to the unused VADDR4 slot -> zeros).
// Element size fixed to 2 bytes (bf16).  3D tile: dim0 elems contiguous,
// dim1 lines with stride0 elements, dim2 planes with stride1 elements.
// Pass tile2=0 for 1D/2D tiles.
// ---------------------------------------------------------------------------
__device__ __forceinline__ void tdm_load_bf16(uint32_t lds_byte_off,
                                              const bf16_t* gptr,
                                              uint32_t tile0, uint32_t tile1,
                                              uint32_t tile2,
                                              uint32_t stride0, uint32_t stride1)
{
    uint64_t ga = (uint64_t)(uintptr_t)gptr;
    v4u g0;
    g0[0] = 1u;                                        // count=1, no gather
    g0[1] = lds_byte_off;                              // lds_addr
    g0[2] = (uint32_t)ga;                              // global_addr[31:0]
    g0[3] = ((uint32_t)(ga >> 32) & 0x01FFFFFFu)       // global_addr[56:32]
            | (2u << 30);                              // type = 2 ("image")
    v8i_t g1;
    g1[0] = (int)(1u << 16);                           // data_size=1 -> 2 bytes
    g1[1] = (int)((tile0 & 0xFFFFu) << 16);            // tensor_dim0[15:0]
    g1[2] = (int)((tile0 >> 16) | ((tile1 & 0xFFFFu) << 16)); // dim0 hi | dim1 lo
    g1[3] = (int)((tile1 >> 16) | (tile0 << 16));      // dim1 hi | tile_dim0
    g1[4] = (int)((tile1 & 0xFFFFu) | ((tile2 & 0xFFFFu) << 16)); // tile_dim1|2
    g1[5] = (int)stride0;                              // tensor_dim0_stride[31:0]
    g1[6] = (int)((stride1 & 0xFFFFu) << 16);          // stride0 hi16 | stride1 lo16
    g1[7] = (int)(stride1 >> 16);                      // stride1[47:16]
    v4i_t g2;
    g2[0] = (int)tile2;                                // tensor_dim2
    g2[1] = 0; g2[2] = 0; g2[3] = 0;
    v4i_t g3;
    g3[0] = 0; g3[1] = 0; g3[2] = 0; g3[3] = 0;
    v8i_t g4;                                          // unused VADDR4 slot
    g4[0] = 0; g4[1] = 0; g4[2] = 0; g4[3] = 0;
    g4[4] = 0; g4[5] = 0; g4[6] = 0; g4[7] = 0;
    __builtin_amdgcn_tensor_load_to_lds(g0, g1, g2, g3, g4, 0);
}

// ---------------------------------------------------------------------------
// Kernel 0: zero GroupNorm stats accumulators and the 2 halo rows (top/bottom)
// of the padded NHWC bf16 image buffer.  (Scratch is not re-poisoned between
// timed replays and kernel 3 accumulates with atomics.)
// ---------------------------------------------------------------------------
__global__ __launch_bounds__(256)
void init_kernel(float* __restrict__ stats, bf16_t* __restrict__ xpad) {
    size_t i = (size_t)blockIdx.x * 256 + threadIdx.x;
    if (i < BATCH * TOPK * NG * 2) stats[i] = 0.0f;
    const size_t nh = (size_t)BATCH * 2 * PADW * CIN;   // 270336 halo elems
    if (i < nh) {
        int cin = (int)(i % CIN);
        int t   = (int)(i / CIN);
        int col = t % PADW;
        int t2  = t / PADW;
        int row = (t2 & 1) ? (PADW - 1) : 0;
        int b   = t2 >> 1;
        xpad[(((size_t)b * PADW + row) * PADW + col) * CIN + cin] = (bf16_t)0.0f;
    }
}

// ---------------------------------------------------------------------------
// Kernel 1: gather + convert SELECTED expert weights to bf16 in the LDS-ready
// layout  wbf[z][g][tap][oc][ic]  (contiguous 18432-byte tiles for the TDM).
// global layout: conv_w[(e*COUT + g*COUTG + oc)][ic][tap]  (tap fastest)
// ---------------------------------------------------------------------------
__global__ __launch_bounds__(256)
void prep_w_kernel(const float* __restrict__ conv_w,
                   const int*   __restrict__ ridx,
                   bf16_t* __restrict__ wbf) {
    size_t i = (size_t)blockIdx.x * 256 + threadIdx.x;  // over 16*8*9216
    int ic  = (int)(i % CING);
    int t   = (int)(i / CING);
    int oc  = t % COUTG;  t /= COUTG;
    int tap = t % 9;      t /= 9;
    int g   = t % NG;
    int z   = t / NG;
    int e   = ridx[z];
    wbf[i] = (bf16_t)conv_w[((size_t)(e * COUT + g * COUTG + oc) * CING + ic) * 9 + tap];
}

// ---------------------------------------------------------------------------
// Kernel 2: convert x (NCHW f32) -> zero-padded NHWC bf16 (8,66,66,256).
// One block per (b,row); LDS transpose (pitch 66 to spread banks) keeps both
// the global read and the global write coalesced.  Bakes the conv halo in so
// the conv kernel's TDM tile loads are dense rectangles.
// ---------------------------------------------------------------------------
__global__ __launch_bounds__(256)
void prep_x_kernel(const float* __restrict__ x, bf16_t* __restrict__ xpad) {
    const int row = blockIdx.x;          // 0..63
    const int b   = blockIdx.y;          // 0..7
    const int tid = threadIdx.x;
    __shared__ bf16_t T[CIN * 66];       // pitch 66 -> stride 33 dwords
    for (int i = tid; i < CIN * WW; i += 256) {
        int cin = i / WW, col = i % WW;  // col fastest: coalesced read
        T[cin * 66 + col] =
            (bf16_t)x[(((size_t)b * CIN + cin) * HH + row) * WW + col];
    }
    __syncthreads();
    bf16_t* orow = xpad + (((size_t)b * PADW + row + 1) * PADW) * CIN;
    for (int i = tid; i < WW * CIN; i += 256) {
        int col = i / CIN, cin = i % CIN;   // cin fastest: coalesced write
        orow[(size_t)(col + 1) * CIN + cin] = T[cin * 66 + col];
    }
    for (int i = tid; i < 2 * CIN; i += 256) {   // halo cols 0 and 65
        int cin = i % CIN;
        int c   = (i < CIN) ? 0 : (PADW - 1);
        orow[(size_t)c * CIN + cin] = (bf16_t)0.0f;
    }
}

// ---------------------------------------------------------------------------
// Kernel 3: grouped 3x3 conv for the selected experts as implicit GEMM on
// v_wmma_f32_16x16x32_bf16.  M = 32 out-channels of one group, N = 256 pixels
// (4 image rows), K = 9 taps x 32 in-channels (one WMMA per tap per tile).
// Tiles are pulled into LDS by the Tensor Data Mover (wave 0 issues two
// tensor_load_to_lds, waits on TENSORcnt, then the block barriers).
// Also produces per-(b,k,group) sum / sum-of-squares for GroupNorm.
// ---------------------------------------------------------------------------
__global__ __launch_bounds__(256)
void conv_wmma_kernel(const bf16_t* __restrict__ xpad,
                      const bf16_t* __restrict__ wbf,
                      float* __restrict__ convbuf,
                      float* __restrict__ stats)
{
    const int tile = blockIdx.x;          // 0..15 : which 4-row band
    const int g    = blockIdx.y;          // 0..7  : conv/GroupNorm group
    const int z    = blockIdx.z;          // 0..15 : b*TOPK + k
    const int b    = z / TOPK;
    const int tid  = threadIdx.x;

    __shared__ alignas(128) bf16_t Wlds[WTILE];           // [tap][oc][ic]
    __shared__ alignas(128) bf16_t Xlds[XR * XC * CING];  // [row][col][ic]
    __shared__ float red[16];

    if (tid < 32) {   // wave 0 drives the TDM; EXEC is ignored by tensor ops
        // weights: 1D tile of 9216 bf16 (18432 B)
        tdm_load_bf16((uint32_t)(uintptr_t)Wlds,
                      wbf + (size_t)(z * NG + g) * WTILE,
                      WTILE, 1, 0, 0, 0);
        // input: 3D tile: 32 ch (contig) x 66 cols (stride 256) x 6 rows
        // (stride 66*256), from the padded NHWC buffer (halo baked in)
        tdm_load_bf16((uint32_t)(uintptr_t)Xlds,
                      xpad + (((size_t)b * PADW + tile * ROWS) * PADW) * CIN
                           + g * CING,
                      CING, XC, XR, CIN, PADW * CIN);
        __builtin_amdgcn_s_wait_tensorcnt(0);
    }
    __syncthreads();

    const int wave = tid >> 5;
    const int lane = tid & 31;
    const int n    = lane & 15;      // row (A) / col (B) within 16
    const int hi   = lane >> 4;      // lane half selects K sub-range

    v8f acc[2][2];                   // [m-tile][n-subtile]
    for (int a = 0; a < 2; ++a)
        for (int c = 0; c < 2; ++c)
            for (int j = 0; j < 8; ++j) acc[a][c][j] = 0.0f;

    for (int tap = 0; tap < 9; ++tap) {
        const int dy = tap / 3, dx = tap - dy * 3;

        // A fragments: 16x32 bf16, lane n = out-channel row; element pairs are
        // consecutive K (=ic): dword reads. K map: i<8 -> hi*8+i, else 16+hi*8+(i-8).
        Frag a0, a1;
        const bf16_t* wt = &Wlds[(tap * COUTG) * CING];
        #pragma unroll
        for (int t = 0; t < 8; ++t) {
            int icb = (t < 4) ? (hi * 8 + 2 * t) : (16 + hi * 8 + 2 * (t - 4));
            a0.u[t] = *(const uint32_t*)&wt[(n     ) * CING + icb];
            a1.u[t] = *(const uint32_t*)&wt[(n + 16) * CING + icb];
        }

        #pragma unroll
        for (int ns = 0; ns < 2; ++ns) {
            // B fragment: 32x16 bf16, lane n = pixel column; K map: i<8 -> hi*8+i, else 16+hi*8+(i-8).
            int p = wave * 32 + ns * 16 + n;   // pixel within 256-pixel tile
            int r = p >> 6;                    // row in tile (0..3)
            int c = p & 63;
            Frag bfr;
            const bf16_t* xt = &Xlds[((r + dy) * XC + (c + dx)) * CING + hi * 8];
            #pragma unroll
            for (int t = 0; t < 8; ++t) bfr.u[t] = *(const uint32_t*)&xt[(t < 4) ? (2 * t) : (16 + 2 * (t - 4))];

            acc[0][ns] = __builtin_amdgcn_wmma_f32_16x16x32_bf16(
                false, a0.v, false, bfr.v, (short)0, acc[0][ns], false, false);
            acc[1][ns] = __builtin_amdgcn_wmma_f32_16x16x32_bf16(
                false, a1.v, false, bfr.v, (short)0, acc[1][ns], false, false);
        }
    }

    // ---- store conv results (f32) + accumulate GroupNorm partial stats.
    // C layout: VGPR j, lanes 0-15 -> M=j, lanes 16-31 -> M=8+j; N = lane&15.
    float s1 = 0.0f, s2 = 0.0f;
    const size_t chbase = (size_t)(z * COUT + g * COUTG);
    const size_t obase  = chbase * HW + (size_t)tile * TILE_PIX;
    #pragma unroll
    for (int m2 = 0; m2 < 2; ++m2) {
        #pragma unroll
        for (int ns = 0; ns < 2; ++ns) {
            int p = wave * 32 + ns * 16 + n;
            #pragma unroll
            for (int j = 0; j < 8; ++j) {
                int m = m2 * 16 + hi * 8 + j;
                float v = acc[m2][ns][j];
                convbuf[obase + (size_t)m * HW + p] = v;
                s1 += v;
                s2 += v * v;
            }
        }
    }
    #pragma unroll
    for (int off = 16; off > 0; off >>= 1) {
        s1 += __shfl_xor(s1, off, 32);
        s2 += __shfl_xor(s2, off, 32);
    }
    if (lane == 0) { red[wave] = s1; red[8 + wave] = s2; }
    __syncthreads();
    if (tid == 0) {
        float t1 = 0.0f, t2 = 0.0f;
        for (int i = 0; i < 8; ++i) { t1 += red[i]; t2 += red[8 + i]; }
        atomicAdd(&stats[(z * NG + g) * 2 + 0], t1);
        atomicAdd(&stats[(z * NG + g) * 2 + 1], t2);
    }
}

// ---------------------------------------------------------------------------
// Kernel 4: GroupNorm (from stats) + per-expert affine + SiLU + top-k combine.
// ---------------------------------------------------------------------------
__global__ __launch_bounds__(256)
void norm_silu_combine_kernel(const float* __restrict__ convbuf,
                              const float* __restrict__ stats,
                              const int*   __restrict__ ridx,
                              const float* __restrict__ rwts,
                              const float* __restrict__ gamma,
                              const float* __restrict__ beta,
                              float* __restrict__ out)
{
    size_t idx = (size_t)blockIdx.x * 256 + threadIdx.x;   // over B*COUT*HW
    int pix = (int)(idx % HW);
    int t   = (int)(idx / HW);
    int oc  = t % COUT;
    int b   = t / COUT;
    int gg  = oc / COUTG;
    const float invN = 1.0f / (float)(COUTG * HW);

    float acc = 0.0f;
    #pragma unroll
    for (int k = 0; k < TOPK; ++k) {
        int z = b * TOPK + k;
        int e = ridx[z];
        float s1  = stats[(z * NG + gg) * 2 + 0];
        float s2  = stats[(z * NG + gg) * 2 + 1];
        float mu  = s1 * invN;
        float var = fmaxf(s2 * invN - mu * mu, 0.0f);
        float rstd = rsqrtf(var + EPSV);
        float v = convbuf[((size_t)z * COUT + oc) * HW + pix];
        float y = (v - mu) * rstd * gamma[e * COUT + oc] + beta[e * COUT + oc];
        float a = y / (1.0f + __expf(-y));   // SiLU
        acc += a * rwts[z];
    }
    out[idx] = acc;
}

// ---------------------------------------------------------------------------
// Launch.  Inputs: (x, routing_weights, routing_indices, top_k, conv_w,
// gamma, beta).  Workspace layout:
//   [convbuf f32 64 MiB][stats 1 KiB][xpad bf16 17.8 MB][wbf bf16 2.4 MB]
// ---------------------------------------------------------------------------
extern "C" void kernel_launch(void* const* d_in, const int* in_sizes, int n_in,
                              void* d_out, int out_size, void* d_ws, size_t ws_size,
                              hipStream_t stream) {
    const float* x      = (const float*)d_in[0];
    const float* rwts   = (const float*)d_in[1];
    const int*   ridx   = (const int*)  d_in[2];
    const float* conv_w = (const float*)d_in[4];
    const float* gamma  = (const float*)d_in[5];
    const float* beta   = (const float*)d_in[6];
    float* out = (float*)d_out;

    char* ws = (char*)d_ws;
    const size_t conv_bytes  = (size_t)BATCH * TOPK * COUT * HW * sizeof(float);
    const size_t stats_bytes = 1024;
    const size_t xpad_bytes  = (size_t)BATCH * PADW * PADW * CIN * sizeof(bf16_t);
    float*  convbuf = (float*)ws;
    float*  stats   = (float*)(ws + conv_bytes);
    bf16_t* xpad    = (bf16_t*)(ws + conv_bytes + stats_bytes);
    bf16_t* wbf     = (bf16_t*)(ws + conv_bytes + stats_bytes + xpad_bytes);

    // 0) zero stats + padded-image halo rows
    init_kernel<<<1056, 256, 0, stream>>>(stats, xpad);

    // 1) gather selected expert weights -> bf16 [z][g][tap][oc][ic]
    prep_w_kernel<<<(BATCH * TOPK * NG * WTILE) / 256, 256, 0, stream>>>(
        conv_w, ridx, wbf);

    // 2) x NCHW f32 -> padded NHWC bf16
    prep_x_kernel<<<dim3(HH, BATCH), 256, 0, stream>>>(x, xpad);

    // 3) WMMA conv for selected experts (+ GroupNorm partial stats)
    dim3 grid(HW / TILE_PIX, NG, BATCH * TOPK);   // (16, 8, 16)
    conv_wmma_kernel<<<grid, 256, 0, stream>>>(xpad, wbf, convbuf, stats);

    // 4) GroupNorm + affine + SiLU + routed combine
    int nblk = (BATCH * COUT * HW) / 256;         // 32768
    norm_silu_combine_kernel<<<nblk, 256, 0, stream>>>(convbuf, stats, ridx,
                                                       rwts, gamma, beta, out);
}
